// Net_25744033972342
// MI455X (gfx1250) — matrix operand order repaired and checked by reading, not compiled, not run on hardware
//
#include <hip/hip_runtime.h>
#include <stdint.h>

#define HDIM   51
#define TLEN   256
#define NBATCH 4096
#define NWAVES 4

typedef __attribute__((ext_vector_type(16))) _Float16 v16h;
typedef __attribute__((ext_vector_type(8)))  float    v8f;

// ---------------- workspace layout (dwords) ----------------
// B-fragments for 3 matrices (W_hh1, W_ih2, W_hh2), each viewed as
// K(64) x N(4 gates * 64) f16 WMMA B-fragments:
//   frag[mat][gate][ut][kstep][lane][vgpr]  (vgpr-contiguous per lane)
// K rows 51/52 carry folded extras:
//   mat0 k=51: W_ih1 (x coefficient), mat0 k=52: b_ih1+b_hh1
//   mat1 k=52: b_ih2+b_hh2  (fed by constant-1 element injected into A k-step 1)
// Sigmoid gates (g=0 i, g=1 f, g=3 o) are pre-scaled by 0.5 so that
// sigmoid(x) = 0.5*tanh(wmma_out) + 0.5 with no extra multiply.
#define FRAG_DW   (3*4*4*2*32*8)     /* 24576 dwords = 96 KB */

#if __has_builtin(__builtin_amdgcn_tanhf)
__device__ __forceinline__ float tanhf_fast(float x) {
    return __builtin_amdgcn_tanhf(x);               // v_tanh_f32 (gfx1250 TRANS op)
}
#else
__device__ __forceinline__ float tanhf_fast(float x) {
    float e = __expf(2.0f * x);
    return 1.0f - 2.0f * __builtin_amdgcn_rcpf(e + 1.0f);
}
#endif
// input already pre-scaled by 0.5 inside the weights
__device__ __forceinline__ float sig_pre(float xhalf) {
    return fmaf(0.5f, tanhf_fast(xhalf), 0.5f);
}

// ============ prep kernel: pack weights into WMMA B-fragments ============
__global__ void lstm_prep(const float* __restrict__ Whh1,
                          const float* __restrict__ Wih2,
                          const float* __restrict__ Whh2,
                          const float* __restrict__ bih1, const float* __restrict__ bhh1,
                          const float* __restrict__ bih2, const float* __restrict__ bhh2,
                          const float* __restrict__ Wih1,
                          uint32_t* __restrict__ ws)
{
    int tid    = blockIdx.x * blockDim.x + threadIdx.x;
    int stride = gridDim.x * blockDim.x;

    for (int idx = tid; idx < FRAG_DW; idx += stride) {
        int r    =  idx        & 7;
        int lane = (idx >> 3)  & 31;
        int ks   = (idx >> 8)  & 1;
        int ut   = (idx >> 9)  & 3;
        int g    = (idx >> 11) & 3;
        int mat  = (idx >> 13);
        const float* W = (mat == 0) ? Whh1 : (mat == 1) ? Wih2 : Whh2;
        // Dense 16-bit B-matrix 32x16 layout: n = lane&15;
        // VGPR 0-3 hold K=8*half+0..7, VGPR 4-7 hold K=16+8*half+0..7; 2 K per VGPR.
        int n    = lane & 15;
        int half = lane >> 4;
        int u    = ut * 16 + n;                 // output unit (N dim)
        int kb   = ks * 32 + half * 8 + 2*r + ((r >= 4) ? 8 : 0);   // input unit (K dim)
        float scale = (g == 2) ? 1.0f : 0.5f;   // pre-scale sigmoid gates
        float f[2];
#pragma unroll
        for (int j = 0; j < 2; ++j) {
            int k = kb + j;
            float v = 0.0f;
            if (u < HDIM) {
                if (k < HDIM)                      v = W[(g*HDIM + u)*HDIM + k];
                else if (mat == 0 && k == HDIM)    v = Wih1[g*HDIM + u];
                else if (mat == 0 && k == HDIM+1)  v = bih1[g*HDIM + u] + bhh1[g*HDIM + u];
                else if (mat == 1 && k == HDIM+1)  v = bih2[g*HDIM + u] + bhh2[g*HDIM + u];
            }
            f[j] = v * scale;
        }
        union { _Float16 h; unsigned short b; } c0, c1;
        c0.h = (_Float16)f[0]; c1.h = (_Float16)f[1];
        ws[idx] = (uint32_t)c0.b | ((uint32_t)c1.b << 16);
    }
}

// ============ main kernel: 1 block = one 16-row batch tile; wave w owns
// hidden-unit tile ut=w (16 units). B-fragments live in registers. ============
__global__ __launch_bounds__(128) void lstm_scan(const float* __restrict__ input,
                                                 const uint32_t* __restrict__ gws,
                                                 const float* __restrict__ Wlin,
                                                 const float* __restrict__ blin,
                                                 float* __restrict__ out)
{
    __shared__ __align__(16) _Float16 h1_lds[16*64];
    __shared__ __align__(16) _Float16 h2_lds[16*64];
    __shared__ __align__(16) float    xs[64*16];
    __shared__ __align__(16) float    pbuf[NWAVES*16];

    const int tid   = threadIdx.x;
    const int wave  = tid >> 5;        // == ut tile owned by this wave
    const int lane  = tid & 31;
    const int laneN = lane & 15;
    const int half  = lane >> 4;
    const int b0    = blockIdx.x * 16;
    const int myut  = wave;

    // ---- hoist this wave's 24 loop-invariant B-fragments into registers ----
    v16h Bf[3][4][2];
#pragma unroll
    for (int mat = 0; mat < 3; ++mat)
#pragma unroll
        for (int g = 0; g < 4; ++g)
#pragma unroll
            for (int ks = 0; ks < 2; ++ks) {
                const uint4* p = (const uint4*)(gws +
                    (size_t)((((mat*4 + g)*4 + myut)*2 + ks)*32 + lane) * 8);
                union { uint4 q[2]; v16h v; } u;
                u.q[0] = p[0]; u.q[1] = p[1];
                Bf[mat][g][ks] = u.v;
            }

    // head weight for this wave's units; head bias
    const int   uu    = myut*16 + laneN;
    const float wlinv = (uu < HDIM) ? Wlin[uu] : 0.0f;
    const float bl    = blin[0];

    // zero h staging; padded units (u>=51) compute h==0 every step (all-zero
    // B columns incl. bias rows), so unconditional stores keep padding intact.
    for (int i = tid; i < 16*64; i += 128) {
        h1_lds[i] = (_Float16)0.0f;
        h2_lds[i] = (_Float16)0.0f;
    }
    float c1[8], c2[8];
#pragma unroll
    for (int r = 0; r < 8; ++r) { c1[r] = 0.0f; c2[r] = 0.0f; }
    __syncthreads();

    // A-fragment (16-bit A 16x32): M = laneN; elems 0..7: K = ks*32+8*half+0..7,
    // elems 8..15: K = ks*32+16+8*half+0..7
    auto loadA = [&](const _Float16* hl, int ks) -> v16h {
        int base = ks * 32 + half * 8;
        union { uint4 q[2]; v16h v; } u;
        u.q[0] = *(const uint4*)(hl + laneN*64 + base);
        u.q[1] = *(const uint4*)(hl + laneN*64 + base + 16);
        return u.v;
    };
    const v8f vzero = {0.f, 0.f, 0.f, 0.f, 0.f, 0.f, 0.f, 0.f};
    const bool h0 = (half == 0);
    _Float16* h1w = h1_lds + half*8*64 + uu;    // this lane's store base
    _Float16* h2w = h2_lds + half*8*64 + uu;

    for (int t = 0; t < TLEN; ++t) {
        if ((t & 63) == 0) {                 // stage 64 timesteps of x (transposed)
            for (int i = tid; i < 64*16; i += 128) {
                int tt = i >> 4, m = i & 15;
                xs[tt*16 + m] = input[(size_t)(b0 + m)*TLEN + (t + tt)];
            }
            __syncthreads();
        }

        // ---- cell 1: gates = [h1_old | x | 1] @ frag(mat0) ----
        v16h a1k0 = loadA(h1_lds, 0);
        v16h a1k1 = loadA(h1_lds, 1);
        {   // branchless injection: x at K=51 (elem 11), 1.0 at K=52 (elem 12)
            _Float16 xv = (_Float16)xs[(t & 63)*16 + laneN];
            a1k1[11] = h0 ? xv : a1k1[11];
            a1k1[12] = h0 ? (_Float16)1.0f : a1k1[12];
        }
        v8f acc[4];
#pragma unroll
        for (int g = 0; g < 4; ++g) {
            acc[g] = __builtin_amdgcn_wmma_f32_16x16x32_f16(false, a1k0, false,
                         Bf[0][g][0], (short)0, vzero, false, false);
            acc[g] = __builtin_amdgcn_wmma_f32_16x16x32_f16(false, a1k1, false,
                         Bf[0][g][1], (short)0, acc[g], false, false);
        }
#pragma unroll
        for (int r = 0; r < 8; ++r) {
            float ii = sig_pre(acc[0][r]);
            float ff = sig_pre(acc[1][r]);
            float gg = tanhf_fast(acc[2][r]);
            float oo = sig_pre(acc[3][r]);
            float c  = ff * c1[r] + ii * gg;
            c1[r] = c;
            h1w[r*64] = (_Float16)(oo * tanhf_fast(c));
        }
        __syncthreads();                     // h1_new visible to all waves

        // ---- cell 2: gates = [h1_new | . | 1] @ frag(mat1) + h2_old @ frag(mat2) ----
        v16h b1k0 = loadA(h1_lds, 0);
        v16h b1k1 = loadA(h1_lds, 1);
        b1k1[12] = h0 ? (_Float16)1.0f : b1k1[12];  // bias feed (K=52)
        v16h a2k0 = loadA(h2_lds, 0);
        v16h a2k1 = loadA(h2_lds, 1);
#pragma unroll
        for (int g = 0; g < 4; ++g) {
            acc[g] = __builtin_amdgcn_wmma_f32_16x16x32_f16(false, b1k0, false,
                         Bf[1][g][0], (short)0, vzero, false, false);
            acc[g] = __builtin_amdgcn_wmma_f32_16x16x32_f16(false, b1k1, false,
                         Bf[1][g][1], (short)0, acc[g], false, false);
            acc[g] = __builtin_amdgcn_wmma_f32_16x16x32_f16(false, a2k0, false,
                         Bf[2][g][0], (short)0, acc[g], false, false);
            acc[g] = __builtin_amdgcn_wmma_f32_16x16x32_f16(false, a2k1, false,
                         Bf[2][g][1], (short)0, acc[g], false, false);
        }
        float part[8];
#pragma unroll
        for (int r = 0; r < 8; ++r) {
            float ii = sig_pre(acc[0][r]);
            float ff = sig_pre(acc[1][r]);
            float gg = tanhf_fast(acc[2][r]);
            float oo = sig_pre(acc[3][r]);
            float c  = ff * c2[r] + ii * gg;
            c2[r] = c;
            float h  = oo * tanhf_fast(c);
            h2w[r*64] = (_Float16)h;
            part[r] = h * wlinv;             // head partial (this lane's unit)
        }

        // head: reduce over 16 lanes (units of this tile), stash per-wave partial
#pragma unroll
        for (int m = 8; m >= 1; m >>= 1)
#pragma unroll
            for (int r = 0; r < 8; ++r) part[r] += __shfl_xor(part[r], m, 32);
        if (laneN == 0) {
#pragma unroll
            for (int r = 0; r < 8; ++r) pbuf[wave*16 + half*8 + r] = part[r];
        }
        __syncthreads();                     // h2_new + partials visible

        // wave 0 combines the 4 per-wave partials (pbuf not rewritten by other
        // waves until they pass the NEXT mid-step barrier, which waits for us)
        if (wave == 0 && lane < 16) {
            float s = pbuf[lane] + pbuf[16 + lane] + pbuf[32 + lane] + pbuf[48 + lane];
            out[(size_t)(b0 + lane)*TLEN + t] = s + bl;
        }
    }
}

extern "C" void kernel_launch(void* const* d_in, const int* in_sizes, int n_in,
                              void* d_out, int out_size, void* d_ws, size_t ws_size,
                              hipStream_t stream) {
    const float* input = (const float*)d_in[0];
    const float* Wih1  = (const float*)d_in[1];
    const float* Whh1  = (const float*)d_in[2];
    const float* bih1  = (const float*)d_in[3];
    const float* bhh1  = (const float*)d_in[4];
    const float* Wih2  = (const float*)d_in[5];
    const float* Whh2  = (const float*)d_in[6];
    const float* bih2  = (const float*)d_in[7];
    const float* bhh2  = (const float*)d_in[8];
    const float* Wlin  = (const float*)d_in[9];
    const float* blin  = (const float*)d_in[10];
    (void)in_sizes; (void)n_in; (void)out_size; (void)ws_size;

    uint32_t* ws = (uint32_t*)d_ws;
    lstm_prep<<<64, 256, 0, stream>>>(Whh1, Wih2, Whh2, bih1, bhh1, bih2, bhh2,
                                      Wih1, ws);
    lstm_scan<<<NBATCH / 16, 128, 0, stream>>>(input, (const uint32_t*)d_ws,
                                               Wlin, blin, (float*)d_out);
}
